// ConExperts_35648228557116
// MI455X (gfx1250) — matrix operand order repaired and checked by reading, not compiled, not run on hardware
//
#include <hip/hip_runtime.h>

// ---------------------------------------------------------------------------
// Grouped MoE FFN for MI455X (gfx1250, wave32, WMMA, async-to-LDS):
//   pass 0: convert x / W1 / W2 fp32 -> bf16 in workspace (memory-bound)
//   pass 1: ffn1 = relu(x_bf16 @ W1_bf16^T)  -> bf16 workspace
//   pass 2: y    = ffn1 @ W2_bf16^T          -> fp32 output
// GEMMs are "NT" (both operands K-contiguous) which maps directly onto the
// CDNA5 WMMA 16x16x32 bf16 fragment layouts with no transpose. Tiles are
// staged into double-buffered LDS with GLOBAL_LOAD_ASYNC_TO_LDS_B128
// (ASYNCcnt pipeline), overlapping DMA of tile k+1 with WMMAs on tile k.
// Workspace requirement: 32MB + 64MB + 64MB + 128MB = 288 MB.
// ---------------------------------------------------------------------------

typedef __bf16 bf16_t;
typedef __attribute__((ext_vector_type(16))) __bf16 v16bf;
typedef __attribute__((ext_vector_type(8)))  __bf16 v8bf;
typedef __attribute__((ext_vector_type(8)))  float  v8f;

#define BM 128
#define BN 128
#define BK 32
#define LDSS 40   // BK + 8 bf16 pad -> 80-byte row stride, conflict-free b128

union Frag16 { v16bf v; v8bf h[2]; };

__device__ __forceinline__ v8bf cvt8(const float4& a, const float4& b) {
    v8bf r;
    r[0] = (__bf16)a.x; r[1] = (__bf16)a.y; r[2] = (__bf16)a.z; r[3] = (__bf16)a.w;
    r[4] = (__bf16)b.x; r[5] = (__bf16)b.y; r[6] = (__bf16)b.z; r[7] = (__bf16)b.w;
    return r;
}

__device__ __forceinline__ void wait_asynccnt0() {
#if defined(__has_builtin)
#if __has_builtin(__builtin_amdgcn_s_wait_asynccnt)
    __builtin_amdgcn_s_wait_asynccnt(0);
    return;
#else
    asm volatile("s_wait_asynccnt 0" ::: "memory");
    return;
#endif
#else
    asm volatile("s_wait_asynccnt 0" ::: "memory");
#endif
}

// Issue async DMA of one 128x32 bf16 tile (8 KB) into padded LDS rows.
// 512 16-byte chunks, 2 per thread (256 threads). Row stride in LDS = 80 B.
__device__ __forceinline__ void issue_tile_async(const bf16_t* __restrict__ g_row0,
                                                 int K, bf16_t* lds_base, int t) {
    #pragma unroll
    for (int c = t; c < 512; c += 256) {
        const int row = c >> 2;        // 0..127
        const int seg = c & 3;         // 16-byte segment within row
        const bf16_t* g = g_row0 + (long)row * K + seg * 8;
        const unsigned lds = (unsigned)(unsigned long long)(lds_base + row * LDSS + seg * 8);
        asm volatile("global_load_async_to_lds_b128 %0, %1, off"
                     :: "v"(lds), "v"(g) : "memory");
    }
}

// ---------------------------------------------------------------------------
// Pass 0: fp32 -> bf16 elementwise (n must be a multiple of 8)
__global__ __launch_bounds__(256)
void f32_to_bf16_kernel(const float* __restrict__ in, bf16_t* __restrict__ out, long n) {
    long i = ((long)blockIdx.x * blockDim.x + threadIdx.x) * 8;
    const long stride = (long)gridDim.x * blockDim.x * 8;
    for (; i < n; i += stride) {
        const float4* p = (const float4*)(in + i);
        float4 a = p[0], b = p[1];
        *(v8bf*)(out + i) = cvt8(a, b);
    }
}

// ---------------------------------------------------------------------------
// D[e, M, N] = act( A[e, M, K] * B[e, N, K]^T ), A/B bf16, double-buffered
// async LDS staging, 8 waves, each wave owns a 64x32 sub-tile (4x2 WMMAs).
template<bool RELU, bool OUT_BF16>
__global__ __launch_bounds__(256)
void grouped_gemm_bf16(const bf16_t* __restrict__ A,
                       const bf16_t* __restrict__ B,
                       void* __restrict__ Dptr,
                       int M, int N, int K)
{
    __shared__ bf16_t sA[2][BM * LDSS];
    __shared__ bf16_t sB[2][BN * LDSS];

    const int e    = blockIdx.z;
    const int m0   = blockIdx.y * BM;
    const int n0   = blockIdx.x * BN;
    const int t    = threadIdx.x;
    const int lane = t & 31;          // wave32
    const int wave = t >> 5;          // 8 waves
    const int half = lane >> 4;
    const int r16  = lane & 15;
    const int mw   = wave & 1;        // 2 x 4 wave grid over the 128x128 tile
    const int nw   = wave >> 1;

    const bf16_t* Ag = A + ((long)e * M + m0) * (long)K;   // row 0 of A tile
    const bf16_t* Bg = B + ((long)e * N + n0) * (long)K;   // row 0 of B tile

    v8f acc[4][2];
    #pragma unroll
    for (int i = 0; i < 4; ++i)
        #pragma unroll
        for (int j = 0; j < 2; ++j)
            acc[i][j] = (v8f){0.f, 0.f, 0.f, 0.f, 0.f, 0.f, 0.f, 0.f};

    const int nk = K / BK;

    // Prologue: DMA tile 0 into buffer 0
    issue_tile_async(Ag, K, &sA[0][0], t);
    issue_tile_async(Bg, K, &sB[0][0], t);
    wait_asynccnt0();
    __syncthreads();

    for (int kt = 0; kt < nk; ++kt) {
        const int cur = kt & 1;
        const int nxt = cur ^ 1;

        // Kick off DMA of the next tile while we compute on the current one.
        if (kt + 1 < nk) {
            issue_tile_async(Ag + (kt + 1) * BK, K, &sA[nxt][0], t);
            issue_tile_async(Bg + (kt + 1) * BK, K, &sB[nxt][0], t);
            if (kt + 2 < nk) {  // warm L2 for the tile after that
                __builtin_prefetch(Ag + (long)(t >> 1) * K + (kt + 2) * BK, 0, 3);
                __builtin_prefetch(Bg + (long)(t >> 1) * K + (kt + 2) * BK, 0, 3);
            }
        }

        // ---- Fragments per CDNA5 16x16x32 bf16 layouts ----
        Frag16 af[4];
        #pragma unroll
        for (int i = 0; i < 4; ++i) {
            const bf16_t* base = &sA[cur][(mw * 64 + i * 16 + r16) * LDSS];
            af[i].h[0] = *(const v8bf*)(base + half * 8);        // K 0..15 half
            af[i].h[1] = *(const v8bf*)(base + 16 + half * 8);   // K 16..31 half
        }
        Frag16 bfr[2];
        #pragma unroll
        for (int j = 0; j < 2; ++j) {
            const bf16_t* base = &sB[cur][(nw * 32 + j * 16 + r16) * LDSS + half * 8];
            bfr[j].h[0] = *(const v8bf*)(base);
            bfr[j].h[1] = *(const v8bf*)(base + 16);
        }

        // ---- 8 WMMAs per wave per K step ----
        #pragma unroll
        for (int i = 0; i < 4; ++i)
            #pragma unroll
            for (int j = 0; j < 2; ++j)
                acc[i][j] = __builtin_amdgcn_wmma_f32_16x16x32_bf16(
                    false, af[i].v, false, bfr[j].v,
                    (short)0, acc[i][j], false, false);

        // Drain this wave's async DMA, then rendezvous so every wave sees
        // the next buffer fully written (and cur is safe to overwrite).
        if (kt + 1 < nk) wait_asynccnt0();
        __syncthreads();
    }

    // ---- Epilogue: C/D layout -> (M = r + 8*half, N = r16) per lane ----
    #pragma unroll
    for (int i = 0; i < 4; ++i) {
        #pragma unroll
        for (int j = 0; j < 2; ++j) {
            const int m = m0 + mw * 64 + i * 16 + half * 8;
            const int n = n0 + nw * 32 + j * 16 + r16;
            long off = ((long)e * M + m) * (long)N + n;
            #pragma unroll
            for (int r = 0; r < 8; ++r) {
                float v = acc[i][j][r];
                if (RELU) v = fmaxf(v, 0.0f);
                if (OUT_BF16) ((bf16_t*)Dptr)[off] = (bf16_t)v;
                else          ((float*)Dptr)[off]  = v;
                off += N;
            }
        }
    }
}

// ---------------------------------------------------------------------------

#define EXPERT_NUM   8
#define D_MODEL      1024
#define D_HIDDEN     4096
#define TOK          2048

extern "C" void kernel_launch(void* const* d_in, const int* in_sizes, int n_in,
                              void* d_out, int out_size, void* d_ws, size_t ws_size,
                              hipStream_t stream) {
    (void)in_sizes; (void)n_in; (void)out_size; (void)ws_size;
    const float* x  = (const float*)d_in[0];   // [8, 2048, 1024]
    const float* w1 = (const float*)d_in[1];   // [8, 4096, 1024]
    const float* w2 = (const float*)d_in[2];   // [8, 1024, 4096]
    float* y = (float*)d_out;                  // [8, 2048, 1024]

    const long nx  = (long)EXPERT_NUM * TOK      * D_MODEL;   // 16.8M
    const long nw1 = (long)EXPERT_NUM * D_HIDDEN * D_MODEL;   // 33.6M
    const long nw2 = (long)EXPERT_NUM * D_MODEL  * D_HIDDEN;  // 33.6M

    // Workspace layout (288 MB total):
    bf16_t* xh   = (bf16_t*)d_ws;            // 32 MB
    bf16_t* w1h  = xh  + nx;                 // 64 MB
    bf16_t* w2h  = w1h + nw1;                // 64 MB
    bf16_t* ffn1 = w2h + nw2;                // 128 MB

    // Pass 0: fp32 -> bf16 (memory-bound, ~23 us at 23.3 TB/s)
    f32_to_bf16_kernel<<<2048, 256, 0, stream>>>(x,  xh,  nx);
    f32_to_bf16_kernel<<<2048, 256, 0, stream>>>(w1, w1h, nw1);
    f32_to_bf16_kernel<<<2048, 256, 0, stream>>>(w2, w2h, nw2);

    // Pass 1 (+ReLU): [e,2048,1024] x [e,4096,1024]^T -> bf16 [e,2048,4096]
    grouped_gemm_bf16<true, true>
        <<<dim3(D_HIDDEN / BN, TOK / BM, EXPERT_NUM), 256, 0, stream>>>(
            xh, w1h, ffn1, TOK, D_HIDDEN, D_MODEL);

    // Pass 2: bf16 [e,2048,4096] x [e,1024,4096]^T -> fp32 [e,2048,1024]
    grouped_gemm_bf16<false, false>
        <<<dim3(D_MODEL / BN, TOK / BM, EXPERT_NUM), 256, 0, stream>>>(
            ffn1, w2h, y, TOK, D_MODEL, D_HIDDEN);
}
